// LSTMBaselineModel_35596688949341
// MI455X (gfx1250) — matrix operand order repaired and checked by reading, not compiled, not run on hardware
//
#include <hip/hip_runtime.h>
#include <hip/hip_bf16.h>

// ===========================================================================
// LSTM (T=512, B=512, IN=300, H=16) + MLP head for MI455X (gfx1250, wave32).
//
// Kernel 1: xg[t*B+b][g] = sum_k x[t,b,k] * W_ih[g,k] + b_ih[g] + b_hh[g]
//   -> 262144x64 GEMM over K=300 using v_wmma_f32_16x16x32_f16 (f32 accum).
//   Memory bound: 315MB read + 64MB write ~ 16us at 23.3 TB/s; WMMA makes
//   the 10.1 GFLOP of math free.
// Kernel 2: 512 sequential LSTM steps + MLP head. thread <-> (row, hidden j),
//   2 rows per wave32; h broadcast via ds_bpermute shuffles (no barriers).
// ===========================================================================

typedef __attribute__((ext_vector_type(16))) _Float16 v16h;
typedef __attribute__((ext_vector_type(8)))  float    v8f;

#define T_STEPS 512
#define BATCH   512
#define IN      300
#define HID     16
#define G4      64          // 4*HID gate width
#define KSTEPS  10          // ceil(300/32): K padded to 320 with zeros

__device__ __forceinline__ float sigf(float x) {
    return 1.0f / (1.0f + __expf(-x));
}
__device__ __forceinline__ float tanh_fast(float x) {
    // tanh(x) = 2*sigmoid(2x) - 1  (single v_exp_f32 + rcp)
    return __builtin_fmaf(2.0f, sigf(2.0f * x), -1.0f);
}

// ---------------------------------------------------------------------------
// Kernel 1: input projection GEMM via WMMA f16 -> f32
// Block = 256 threads = 8 waves; each wave owns a 16(M) x 64(N) strip.
// Block M-tile = 128 rows; grid = 262144/128 = 2048 blocks.
// ---------------------------------------------------------------------------
__global__ __launch_bounds__(256) void lstm_inproj_wmma(
    const float* __restrict__ x,      // [T*B, 300]
    const float* __restrict__ W_ih,   // [64, 300]
    const float* __restrict__ b_ih,   // [64]
    const float* __restrict__ b_hh,   // [64]
    float* __restrict__ xg)           // [T*B, 64]
{
    // W_ih^T staged as ready-made per-lane B fragments: [kstep][ntile][lane][16]
    __shared__ _Float16 Bsh[KSTEPS * 4 * 32 * 16];   // 40 KB

    const int tid  = threadIdx.x;
    const int lane = tid & 31;
    const int wave = tid >> 5;

    // Fill B fragments (B[k][n] = W_ih[n][k]; zero-pad k >= 300).
    // v16h element i of lane l holds B[kstep*32 + (i&7) + (i>=8 ? 16:0) + (l>=16 ? 8:0)][ntile*16 + (l&15)]
    for (int e = tid; e < KSTEPS * 4 * 32 * 16; e += 256) {
        const int i  = e & 15;
        const int ln = (e >> 4) & 31;
        const int nt = (e >> 9) & 3;
        const int ks = e >> 11;
        const int k  = ks * 32 + (i & 7) + ((i & 8) ? 16 : 0) + ((ln & 16) ? 8 : 0);
        const int n  = nt * 16 + (ln & 15);
        Bsh[e] = (k < IN) ? (_Float16)W_ih[n * IN + k] : (_Float16)0.0f;
    }
    __syncthreads();

    const int m0      = blockIdx.x * 128 + wave * 16;
    const int laneoff = (lane >> 4) & 1;                 // upper half-wave flag
    const float* xrow = x + (size_t)(m0 + (lane & 15)) * IN;

    v8f acc[4];
    #pragma unroll
    for (int n = 0; n < 4; ++n)
        acc[n] = (v8f){0.f, 0.f, 0.f, 0.f, 0.f, 0.f, 0.f, 0.f};

    for (int ks = 0; ks < KSTEPS; ++ks) {
        // A fragment (16-bit A 16x32 layout): lane l, element i holds
        // x[m0 + (l&15)][ks*32 + i + (i>=8 ? 8:0) + (l>=16 ? 8:0)]
        v16h a;
        const int kb = ks * 32 + laneoff * 8;
        if (ks < KSTEPS - 1) {
            // fully in range: two contiguous 8-float runs, 16B-aligned
            const float4 p0 = *(const float4*)(xrow + kb);
            const float4 p1 = *(const float4*)(xrow + kb + 4);
            const float4 p2 = *(const float4*)(xrow + kb + 16);
            const float4 p3 = *(const float4*)(xrow + kb + 20);
            a[0]  = (_Float16)p0.x; a[1]  = (_Float16)p0.y;
            a[2]  = (_Float16)p0.z; a[3]  = (_Float16)p0.w;
            a[4]  = (_Float16)p1.x; a[5]  = (_Float16)p1.y;
            a[6]  = (_Float16)p1.z; a[7]  = (_Float16)p1.w;
            a[8]  = (_Float16)p2.x; a[9]  = (_Float16)p2.y;
            a[10] = (_Float16)p2.z; a[11] = (_Float16)p2.w;
            a[12] = (_Float16)p3.x; a[13] = (_Float16)p3.y;
            a[14] = (_Float16)p3.z; a[15] = (_Float16)p3.w;
        } else {
            // K tail 288..319: guard k < 300
            #pragma unroll
            for (int i = 0; i < 16; ++i) {
                const int kl = kb + i + ((i >= 8) ? 8 : 0);
                a[i] = (kl < IN) ? (_Float16)xrow[kl] : (_Float16)0.0f;
            }
        }
        #pragma unroll
        for (int n = 0; n < 4; ++n) {
            const v16h b = *(const v16h*)&Bsh[((ks * 4 + n) * 32 + lane) * 16];
            acc[n] = __builtin_amdgcn_wmma_f32_16x16x32_f16(
                false, a, false, b, (short)0, acc[n], false, false);
        }
    }

    // C layout: VGPR r, lanes 0-15 -> (M=r, N=lane); lanes 16-31 -> (M=8+r, N=lane-16)
    #pragma unroll
    for (int n = 0; n < 4; ++n) {
        const int col  = n * 16 + (lane & 15);
        const float bb = b_ih[col] + b_hh[col];
        float* outp = xg + (size_t)(m0 + laneoff * 8) * G4 + col;
        #pragma unroll
        for (int r = 0; r < 8; ++r)
            outp[(size_t)r * G4] = acc[n][r] + bb;
    }
}

// ---------------------------------------------------------------------------
// Kernel 2: sequential recurrence + fused MLP head.
// Block = 256 threads = 8 waves = 16 batch rows; grid = 32 blocks.
// Thread (row, j): owns c[row][j], computes gates i/f/g/o for hidden unit j.
// h shared inside the 16-lane row group by shuffles -> no barriers, no LDS.
// ---------------------------------------------------------------------------
__global__ __launch_bounds__(256) void lstm_recur_head(
    const float* __restrict__ xg,    // [T, B, 64] (biases already added)
    const float* __restrict__ W_hh,  // [64, 16]
    const float* __restrict__ W1,    // [64, 16]
    const float* __restrict__ b1,    // [64]
    const float* __restrict__ W2,    // [1, 64]
    const float* __restrict__ b2,    // [1]
    float* __restrict__ out)         // [B]
{
    const int lane = threadIdx.x & 31;
    const int wave = threadIdx.x >> 5;
    const int j    = lane & 15;
    const int row  = blockIdx.x * 16 + wave * 2 + (lane >> 4);

    // Recurrent weight rows for this thread's 4 gates (torch order i,f,g,o)
    float Wi[HID], Wf[HID], Wg[HID], Wo[HID];
    #pragma unroll
    for (int k = 0; k < HID; ++k) {
        Wi[k] = W_hh[(j     ) * HID + k];
        Wf[k] = W_hh[(j + 16) * HID + k];
        Wg[k] = W_hh[(j + 32) * HID + k];
        Wo[k] = W_hh[(j + 48) * HID + k];
    }

    float hvec[HID];
    #pragma unroll
    for (int k = 0; k < HID; ++k) hvec[k] = 0.0f;
    float c = 0.0f;

    const float*  xp      = xg + (size_t)row * G4 + j;
    const size_t  tstride = (size_t)BATCH * G4;

    // software pipeline: keep next step's xg in flight over current compute
    float cgi = xp[0], cgf = xp[16], cgg = xp[32], cgo = xp[48];

    for (int t = 0; t < T_STEPS; ++t) {
        float ngi = 0.f, ngf = 0.f, ngg = 0.f, ngo = 0.f;
        if (t + 1 < T_STEPS) {
            const float* np = xp + (size_t)(t + 1) * tstride;
            ngi = np[0]; ngf = np[16]; ngg = np[32]; ngo = np[48];
        }

        float gi = cgi, gf = cgf, gg = cgg, go = cgo;
        #pragma unroll
        for (int k = 0; k < HID; ++k) {
            const float hk = hvec[k];
            gi = __builtin_fmaf(hk, Wi[k], gi);
            gf = __builtin_fmaf(hk, Wf[k], gf);
            gg = __builtin_fmaf(hk, Wg[k], gg);
            go = __builtin_fmaf(hk, Wo[k], go);
        }

        c = sigf(gf) * c + sigf(gi) * tanh_fast(gg);
        const float hj = sigf(go) * tanh_fast(c);

        // broadcast updated h across this row's 16 lanes (ds_bpermute)
        #pragma unroll
        for (int k = 0; k < HID; ++k)
            hvec[k] = __shfl(hj, (lane & 16) | k, 32);

        cgi = ngi; cgf = ngf; cgg = ngg; cgo = ngo;
    }

    // MLP head: z = relu(h @ W1^T + b1); out = 4*sigmoid(z @ W2^T + b2)
    float partial = 0.0f;
    #pragma unroll
    for (int q = 0; q < 4; ++q) {
        const int o = q * 16 + j;
        float z = b1[o];
        #pragma unroll
        for (int k = 0; k < HID; ++k)
            z = __builtin_fmaf(hvec[k], W1[o * HID + k], z);
        z = fmaxf(z, 0.0f);
        partial = __builtin_fmaf(z, W2[o], partial);
    }
    // reduce across the 16 lanes of this row (xor masks stay inside a half)
    partial += __shfl_xor(partial, 1, 32);
    partial += __shfl_xor(partial, 2, 32);
    partial += __shfl_xor(partial, 4, 32);
    partial += __shfl_xor(partial, 8, 32);

    if (j == 0) out[row] = 4.0f * sigf(partial + b2[0]);
}

// ---------------------------------------------------------------------------
extern "C" void kernel_launch(void* const* d_in, const int* in_sizes, int n_in,
                              void* d_out, int out_size, void* d_ws, size_t ws_size,
                              hipStream_t stream) {
    (void)in_sizes; (void)n_in; (void)out_size; (void)ws_size;
    const float* x    = (const float*)d_in[0];
    const float* W_ih = (const float*)d_in[1];
    const float* W_hh = (const float*)d_in[2];
    const float* b_ih = (const float*)d_in[3];
    const float* b_hh = (const float*)d_in[4];
    const float* W1   = (const float*)d_in[5];
    const float* b1   = (const float*)d_in[6];
    const float* W2   = (const float*)d_in[7];
    const float* b2   = (const float*)d_in[8];
    float* out = (float*)d_out;
    float* xg  = (float*)d_ws;   // needs T*B*64*4 = 64 MiB of workspace

    lstm_inproj_wmma<<<dim3((T_STEPS * BATCH) / 128), dim3(256), 0, stream>>>(
        x, W_ih, b_ih, b_hh, xg);
    lstm_recur_head<<<dim3(BATCH / 16), dim3(256), 0, stream>>>(
        xg, W_hh, W1, b1, W2, b2, out);
}
